// CustomMultiHeadAttention_53884659695679
// MI455X (gfx1250) — matrix operand order repaired and checked by reading, not compiled, not run on hardware
//
#include <hip/hip_runtime.h>
#include <hip/hip_bf16.h>

#define DEV __device__ __forceinline__

typedef __attribute__((ext_vector_type(16))) __bf16          v16bf;
typedef __attribute__((ext_vector_type(8)))  float           v8f;
typedef __attribute__((ext_vector_type(16))) unsigned short  v16u;
typedef __attribute__((ext_vector_type(8)))  unsigned short  v8u;

constexpr int   Bn  = 4;
constexpr int   Sn  = 2560;
constexpr int   Dn  = 2048;
constexpr int   Hn  = 16;
constexpr int   DKn = 128;
constexpr int   Pn  = 2048;
constexpr int   Cn  = 512;
// softmax scale 1/sqrt(128) pre-multiplied by log2(e): scores live in base-2 domain,
// so the flash softmax is a single native v_exp_f32 per element.
constexpr float QSCALE = 0.08838834764831843f * 1.4426950408889634f;

// ---------- helpers ----------

DEV unsigned short f2bf(float f) {           // round-to-nearest-even f32 -> bf16
    unsigned u = __float_as_uint(f);
    u += 0x7FFFu + ((u >> 16) & 1u);
    return (unsigned short)(u >> 16);
}
DEV float bf2f(unsigned short s) { return __uint_as_float(((unsigned)s) << 16); }

DEV float fast_exp2(float x) { return __builtin_amdgcn_exp2f(x); }  // raw v_exp_f32

DEV v8f wmma_bf16(v16bf a, v16bf b, v8f c) {
    return __builtin_amdgcn_wmma_f32_16x16x32_bf16(false, a, false, b, (short)0, c,
                                                   false, false);
}

// A-fragment (16x32 bf16, M=lane&15) from bf16 row-major (global or LDS).
DEV v16bf a_from_bf16(const unsigned short* base, int ld, int row, int kb, int lane) {
    int half = lane >> 4;
    const unsigned short* p = base + (size_t)row * ld + kb + 16 * half;
    v16u u = *(const v16u*)p;       // K = 16*half + j
    return __builtin_bit_cast(v16bf, u);
}

// B-fragment (32x16 bf16, N=lane&15, K contiguous) from bf16 row-major [col][k].
DEV v16bf b_from_bf16(const unsigned short* __restrict__ base, int ld, int col, int kb,
                      int lane) {
    const unsigned short* p = base + (size_t)col * ld + kb + 16 * (lane >> 4);
    v16u u = *(const v16u*)p;       // one 32B load per lane
    return __builtin_bit_cast(v16bf, u);
}

DEV float red_max16(float v) {
#pragma unroll
    for (int m = 1; m < 16; m <<= 1) v = fmaxf(v, __shfl_xor(v, m, 32));
    return v;
}
DEV float red_sum16(float v) {
#pragma unroll
    for (int m = 1; m < 16; m <<= 1) v += __shfl_xor(v, m, 32);
    return v;
}

// ---------- kernel 0: one-shot f32 -> bf16 conversion (memory-bound) ----------
__global__ __launch_bounds__(256) void cvt_bf16_kernel(const float* __restrict__ src,
                                                       unsigned short* __restrict__ dst,
                                                       long long n) {
    long long i = ((long long)blockIdx.x * 256 + threadIdx.x) * 8;
    if (i + 8 <= n) {
        v8u o;
#pragma unroll
        for (int j = 0; j < 8; ++j) o[j] = f2bf(src[i + j]);
        *(v8u*)(dst + i) = o;
    }
}

// ---------- kernel 1: projection GEMM  Y = (X @ W^T + b) * oscale  (bf16) ----------
// vt==0: store as (B,H,S,DK).  vt==1: store transposed (B,H,DK,S) (for V).
// oscale folds the attention softmax scale (and log2e) into Q at zero cost.
__global__ __launch_bounds__(128) void qkv_proj_kernel(
    const unsigned short* __restrict__ x, const unsigned short* __restrict__ W,
    const float* __restrict__ bias, unsigned short* __restrict__ dst, int vt,
    float oscale) {
    int mt = blockIdx.x * 64, nt = blockIdx.y * 64;
    int wave = threadIdx.x >> 5, lane = threadIdx.x & 31;
    int hi = lane >> 4, ln = lane & 15;
    int mrow = mt + wave * 16 + ln;

    v8f acc[4];
#pragma unroll
    for (int t = 0; t < 4; ++t)
#pragma unroll
        for (int i = 0; i < 8; ++i) acc[t][i] = 0.f;

    for (int kb = 0; kb < Dn; kb += 32) {
        v16bf a = a_from_bf16(x, Dn, mrow, kb, lane);
#pragma unroll
        for (int t = 0; t < 4; ++t) {
            v16bf bw = b_from_bf16(W, Dn, nt + t * 16 + ln, kb, lane);
            acc[t] = wmma_bf16(a, bw, acc[t]);
        }
    }
#pragma unroll
    for (int t = 0; t < 4; ++t) {
        int n = nt + t * 16 + ln;
        int h = n / DKn, dk = n % DKn;
        float bv = bias[n];
#pragma unroll
        for (int i = 0; i < 8; ++i) {
            int m = mt + wave * 16 + i + 8 * hi;
            int b = m / Sn, s = m % Sn;
            float v = (acc[t][i] + bv) * oscale;
            size_t idx = vt ? ((((size_t)b * Hn + h) * DKn + dk) * Sn + s)
                            : ((((size_t)b * Hn + h) * Sn + s) * DKn + dk);
            dst[idx] = f2bf(v);
        }
    }
}

// ---------- kernel 2: causal prefix flash attention (scores already base-2) ----------
__global__ __launch_bounds__(128) void prefix_attn_kernel(
    const unsigned short* __restrict__ Q, const unsigned short* __restrict__ K,
    const unsigned short* __restrict__ Vt, unsigned short* __restrict__ ctx) {
    __shared__ unsigned short Pl[4][16 * 32];
    int bh = blockIdx.y;
    int b = bh / Hn, h = bh % Hn;
    int wave = threadIdx.x >> 5, lane = threadIdx.x & 31;
    int hi = lane >> 4, ln = lane & 15;
    int qbase = blockIdx.x * 64 + wave * 16;

    const unsigned short* Qh = Q + (size_t)bh * Sn * DKn;
    const unsigned short* Kh = K + (size_t)bh * Sn * DKn;
    const unsigned short* Vh = Vt + (size_t)bh * DKn * Sn;
    unsigned short* pw = Pl[wave];

    v16bf qf[4];
#pragma unroll
    for (int dc = 0; dc < 4; ++dc) qf[dc] = a_from_bf16(Qh, DKn, qbase + ln, dc * 32, lane);

    v8f O[8];
    float mst[8], lst[8];
#pragma unroll
    for (int t = 0; t < 8; ++t)
#pragma unroll
        for (int i = 0; i < 8; ++i) O[t][i] = 0.f;
#pragma unroll
    for (int i = 0; i < 8; ++i) { mst[i] = -1e30f; lst[i] = 0.f; }

    int nkb = (qbase + 16 + 31) >> 5;  // causal: key blocks covering k <= qbase+15
    for (int kb = 0; kb < nkb; ++kb) {
        int k0 = kb * 32;
        // prefetch next key block's K rows and V columns while this block computes
        if (kb + 1 < nkb) {
            __builtin_prefetch(Kh + (size_t)(k0 + 32 + ln) * DKn, 0, 1);
            __builtin_prefetch(Kh + (size_t)(k0 + 48 + ln) * DKn, 0, 1);
            __builtin_prefetch(Vh + (size_t)(lane * 4) * Sn + k0 + 32, 0, 1);
        }
        v8f s0, s1;
#pragma unroll
        for (int i = 0; i < 8; ++i) { s0[i] = 0.f; s1[i] = 0.f; }
#pragma unroll
        for (int dc = 0; dc < 4; ++dc) {
            v16bf bk0 = b_from_bf16(Kh, DKn, k0 + ln, dc * 32, lane);
            v16bf bk1 = b_from_bf16(Kh, DKn, k0 + 16 + ln, dc * 32, lane);
            s0 = wmma_bf16(qf[dc], bk0, s0);
            s1 = wmma_bf16(qf[dc], bk1, s1);
        }
        // causal mask + online softmax (base-2); P tile -> LDS (bf16)
#pragma unroll
        for (int i = 0; i < 8; ++i) {
            int q = qbase + i + 8 * hi;
            float x0 = s0[i], x1 = s1[i];
            if (k0 + ln > q) x0 = -1e30f;
            if (k0 + 16 + ln > q) x1 = -1e30f;
            float rm = red_max16(fmaxf(x0, x1));
            float mn = fmaxf(mst[i], rm);
            float corr = fast_exp2(mst[i] - mn);
            float p0 = fast_exp2(x0 - mn), p1 = fast_exp2(x1 - mn);
            float rs = red_sum16(p0 + p1);
            lst[i] = lst[i] * corr + rs;
            mst[i] = mn;
#pragma unroll
            for (int t = 0; t < 8; ++t) O[t][i] *= corr;
            pw[(i + 8 * hi) * 32 + ln] = f2bf(p0);
            pw[(i + 8 * hi) * 32 + 16 + ln] = f2bf(p1);
        }
        // P (A-frag) x V^T (B-frags, contiguous thanks to (B,H,DK,S) layout)
        v16bf pa = a_from_bf16(pw, 32, ln, 0, lane);
#pragma unroll
        for (int t = 0; t < 8; ++t) {
            v16bf bv = b_from_bf16(Vh, Sn, t * 16 + ln, k0, lane);
            O[t] = wmma_bf16(pa, bv, O[t]);
        }
    }
    // finalize: O/l -> ctx (B*S, D) bf16, merged heads
#pragma unroll
    for (int i = 0; i < 8; ++i) {
        float inv = 1.0f / lst[i];
        int row = qbase + i + 8 * hi;
        size_t ro = ((size_t)b * Sn + row) * Dn + h * DKn;
#pragma unroll
        for (int t = 0; t < 8; ++t) ctx[ro + t * 16 + ln] = f2bf(O[t][i] * inv);
    }
}

// ---------- kernel 3: candidate attention (prefix keys + self only) ----------
__global__ __launch_bounds__(128) void cand_attn_kernel(
    const unsigned short* __restrict__ Q, const unsigned short* __restrict__ K,
    const unsigned short* __restrict__ Vt, unsigned short* __restrict__ ctx) {
    __shared__ unsigned short Pl[4][16 * 32];
    int bh = blockIdx.y;
    int b = bh / Hn, h = bh % Hn;
    int wave = threadIdx.x >> 5, lane = threadIdx.x & 31;
    int hi = lane >> 4, ln = lane & 15;
    int cbase = blockIdx.x * 64 + wave * 16;   // 0..C-16
    int qrow0 = Pn + cbase;                    // global query row base

    const unsigned short* Qh = Q + (size_t)bh * Sn * DKn;
    const unsigned short* Kh = K + (size_t)bh * Sn * DKn;
    const unsigned short* Vh = Vt + (size_t)bh * DKn * Sn;
    unsigned short* pw = Pl[wave];

    v16bf qf[4];
#pragma unroll
    for (int dc = 0; dc < 4; ++dc) qf[dc] = a_from_bf16(Qh, DKn, qrow0 + ln, dc * 32, lane);

    v8f O[8];
    float mst[8], lst[8];
#pragma unroll
    for (int t = 0; t < 8; ++t)
#pragma unroll
        for (int i = 0; i < 8; ++i) O[t][i] = 0.f;
#pragma unroll
    for (int i = 0; i < 8; ++i) { mst[i] = -1e30f; lst[i] = 0.f; }

    for (int kb = 0; kb < Pn / 32; ++kb) {     // full prefix, no mask
        int k0 = kb * 32;
        if (kb + 1 < Pn / 32) {
            __builtin_prefetch(Kh + (size_t)(k0 + 32 + ln) * DKn, 0, 1);
            __builtin_prefetch(Kh + (size_t)(k0 + 48 + ln) * DKn, 0, 1);
            __builtin_prefetch(Vh + (size_t)(lane * 4) * Sn + k0 + 32, 0, 1);
        }
        v8f s0, s1;
#pragma unroll
        for (int i = 0; i < 8; ++i) { s0[i] = 0.f; s1[i] = 0.f; }
#pragma unroll
        for (int dc = 0; dc < 4; ++dc) {
            v16bf bk0 = b_from_bf16(Kh, DKn, k0 + ln, dc * 32, lane);
            v16bf bk1 = b_from_bf16(Kh, DKn, k0 + 16 + ln, dc * 32, lane);
            s0 = wmma_bf16(qf[dc], bk0, s0);
            s1 = wmma_bf16(qf[dc], bk1, s1);
        }
#pragma unroll
        for (int i = 0; i < 8; ++i) {
            float x0 = s0[i], x1 = s1[i];
            float rm = red_max16(fmaxf(x0, x1));
            float mn = fmaxf(mst[i], rm);
            float corr = fast_exp2(mst[i] - mn);
            float p0 = fast_exp2(x0 - mn), p1 = fast_exp2(x1 - mn);
            float rs = red_sum16(p0 + p1);
            lst[i] = lst[i] * corr + rs;
            mst[i] = mn;
#pragma unroll
            for (int t = 0; t < 8; ++t) O[t][i] *= corr;
            pw[(i + 8 * hi) * 32 + ln] = f2bf(p0);
            pw[(i + 8 * hi) * 32 + 16 + ln] = f2bf(p1);
        }
        v16bf pa = a_from_bf16(pw, 32, ln, 0, lane);
#pragma unroll
        for (int t = 0; t < 8; ++t) {
            v16bf bv = b_from_bf16(Vh, Sn, t * 16 + ln, k0, lane);
            O[t] = wmma_bf16(pa, bv, O[t]);
        }
    }

    // self term: diagonal of Qc*Kc^T tile (4 WMMAs); Q already carries the scale.
    v8f sc;
#pragma unroll
    for (int i = 0; i < 8; ++i) sc[i] = 0.f;
#pragma unroll
    for (int dc = 0; dc < 4; ++dc) {
        v16bf bkc = b_from_bf16(Kh, DKn, qrow0 + ln, dc * 32, lane);
        sc = wmma_bf16(qf[dc], bkc, sc);
    }
#pragma unroll
    for (int i = 0; i < 8; ++i) {
        // diag row r: r<8 at lane r slot r; r>=8 at lane r+16 slot r-8.
        int src = hi ? (i + 24) : i;
        float sself = __shfl(sc[i], src, 32);
        float mn = fmaxf(mst[i], sself);
        float corr = fast_exp2(mst[i] - mn);
        float ps = fast_exp2(sself - mn);
        lst[i] = lst[i] * corr + ps;
        mst[i] = mn;
        int row = qrow0 + i + 8 * hi;
#pragma unroll
        for (int t = 0; t < 8; ++t) {
            float vv = bf2f(Vh[(size_t)(t * 16 + ln) * Sn + row]);
            O[t][i] = O[t][i] * corr + ps * vv;
        }
    }
#pragma unroll
    for (int i = 0; i < 8; ++i) {
        float inv = 1.0f / lst[i];
        int row = qrow0 + i + 8 * hi;
        size_t ro = ((size_t)b * Sn + row) * Dn + h * DKn;
#pragma unroll
        for (int t = 0; t < 8; ++t) ctx[ro + t * 16 + ln] = f2bf(O[t][i] * inv);
    }
}

// ---------- kernel 4: output projection  out = ctx @ Wo^T + bo (f32 out) ----------
__global__ __launch_bounds__(128) void out_proj_kernel(
    const unsigned short* __restrict__ ctx, const unsigned short* __restrict__ Wo,
    const float* __restrict__ bo, float* __restrict__ out) {
    int mt = blockIdx.x * 64, nt = blockIdx.y * 64;
    int wave = threadIdx.x >> 5, lane = threadIdx.x & 31;
    int hi = lane >> 4, ln = lane & 15;
    int mrow = mt + wave * 16 + ln;

    v8f acc[4];
#pragma unroll
    for (int t = 0; t < 4; ++t)
#pragma unroll
        for (int i = 0; i < 8; ++i) acc[t][i] = 0.f;

    for (int kb = 0; kb < Dn; kb += 32) {
        v16bf a = a_from_bf16(ctx, Dn, mrow, kb, lane);
#pragma unroll
        for (int t = 0; t < 4; ++t) {
            v16bf bw = b_from_bf16(Wo, Dn, nt + t * 16 + ln, kb, lane);
            acc[t] = wmma_bf16(a, bw, acc[t]);
        }
    }
#pragma unroll
    for (int t = 0; t < 4; ++t) {
        int n = nt + t * 16 + ln;
        float bv = bo[n];
#pragma unroll
        for (int i = 0; i < 8; ++i) {
            int m = mt + wave * 16 + i + 8 * hi;
            out[(size_t)m * Dn + n] = acc[t][i] + bv;
        }
    }
}

// ---------- launcher ----------
extern "C" void kernel_launch(void* const* d_in, const int* in_sizes, int n_in,
                              void* d_out, int out_size, void* d_ws, size_t ws_size,
                              hipStream_t stream) {
    const float* query = (const float*)d_in[0];
    const float* key   = (const float*)d_in[1];
    const float* value = (const float*)d_in[2];
    const float* Wq = (const float*)d_in[3];
    const float* bq = (const float*)d_in[4];
    const float* Wk = (const float*)d_in[5];
    const float* bk = (const float*)d_in[6];
    const float* Wv = (const float*)d_in[7];
    const float* bv = (const float*)d_in[8];
    const float* Wo = (const float*)d_in[9];
    const float* bo = (const float*)d_in[10];
    float* out = (float*)d_out;

    const size_t NX = (size_t)Bn * Sn * Dn;  // 20,971,520 (activations / ctx)
    const size_t NW = (size_t)Dn * Dn;       //  4,194,304 (weights)

    unsigned short* p   = (unsigned short*)d_ws;
    unsigned short* Qb  = p; p += NX;
    unsigned short* Kb  = p; p += NX;
    unsigned short* Vtb = p; p += NX;
    unsigned short* ctx = p; p += NX;
    unsigned short* Xq  = p; p += NX;
    unsigned short* Xk  = p; p += NX;
    unsigned short* Xv  = p; p += NX;
    unsigned short* Wqb = p; p += NW;
    unsigned short* Wkb = p; p += NW;
    unsigned short* Wvb = p; p += NW;
    unsigned short* Wob = p; p += NW;

    dim3 cblk(256);
    cvt_bf16_kernel<<<dim3(NX / 2048), cblk, 0, stream>>>(query, Xq, (long long)NX);
    cvt_bf16_kernel<<<dim3(NX / 2048), cblk, 0, stream>>>(key, Xk, (long long)NX);
    cvt_bf16_kernel<<<dim3(NX / 2048), cblk, 0, stream>>>(value, Xv, (long long)NX);
    cvt_bf16_kernel<<<dim3(NW / 2048), cblk, 0, stream>>>(Wq, Wqb, (long long)NW);
    cvt_bf16_kernel<<<dim3(NW / 2048), cblk, 0, stream>>>(Wk, Wkb, (long long)NW);
    cvt_bf16_kernel<<<dim3(NW / 2048), cblk, 0, stream>>>(Wv, Wvb, (long long)NW);
    cvt_bf16_kernel<<<dim3(NW / 2048), cblk, 0, stream>>>(Wo, Wob, (long long)NW);

    dim3 blk(128);
    dim3 gp((Bn * Sn) / 64, Dn / 64);  // 160 x 32

    // Q carries 1/sqrt(dk) * log2(e); K and V are unscaled.
    qkv_proj_kernel<<<gp, blk, 0, stream>>>(Xq, Wqb, bq, Qb, 0, QSCALE);
    qkv_proj_kernel<<<gp, blk, 0, stream>>>(Xk, Wkb, bk, Kb, 0, 1.0f);
    qkv_proj_kernel<<<gp, blk, 0, stream>>>(Xv, Wvb, bv, Vtb, 1, 1.0f);  // V transposed

    prefix_attn_kernel<<<dim3(Pn / 64, Bn * Hn), blk, 0, stream>>>(Qb, Kb, Vtb, ctx);
    cand_attn_kernel<<<dim3(Cn / 64, Bn * Hn), blk, 0, stream>>>(Qb, Kb, Vtb, ctx);

    out_proj_kernel<<<gp, blk, 0, stream>>>(ctx, Wob, bo, out);
}
